// Temp_Attn_23244363006106
// MI455X (gfx1250) — matrix operand order repaired and checked by reading, not compiled, not run on hardware
//
#include <hip/hip_runtime.h>

typedef __attribute__((ext_vector_type(16))) __bf16 v16bf;
typedef __attribute__((ext_vector_type(8)))  float  v8f;

#define NB  512
#define NT  32
#define NF  4096
#define NL  256
#define NBT (NB*NT)      // 16384
#define NK3 (3*NF)       // 12288

#if defined(__gfx1250__) && __has_builtin(__builtin_amdgcn_global_load_async_to_lds_b128) && __has_builtin(__builtin_amdgcn_s_wait_asynccnt)
#define USE_ASYNC_LDS 1
#define AS1 __attribute__((address_space(1)))
#define AS3 __attribute__((address_space(3)))
typedef int i4v __attribute__((vector_size(16)));
#else
#define USE_ASYNC_LDS 0
#endif

__device__ __forceinline__ __bf16 tobf(float f) {
  union { float f; unsigned u; } a; a.f = f;
  unsigned r = (a.u + 0x7fffu + ((a.u >> 16) & 1u)) >> 16;   // RNE truncate
  union { unsigned short s; __bf16 b; } o; o.s = (unsigned short)r;
  return o.b;
}
__device__ __forceinline__ float nanfix(float v) { return (v != v) ? 0.0f : v; }
__device__ __forceinline__ float sigm(float v)   { return 1.0f / (1.0f + __expf(-v)); }

// ---------- one-pass f32 -> bf16 cast (with NaN scrub for x) ----------
template<int FIXNAN>
__global__ void cast_f2b_kernel(const float* __restrict__ src, __bf16* __restrict__ dst, int n) {
  int i = (blockIdx.x * 256 + threadIdx.x) * 4;
  if (i >= n) return;
  float4 f = *(const float4*)(src + i);
  if (FIXNAN) { f.x = nanfix(f.x); f.y = nanfix(f.y); f.z = nanfix(f.z); f.w = nanfix(f.w); }
  dst[i] = tobf(f.x); dst[i + 1] = tobf(f.y); dst[i + 2] = tobf(f.z); dst[i + 3] = tobf(f.w);
}

// ---------- weight packing: tsn_w (L, F_IN, 3) -> Wtb[l][s*4096+c] (bf16, K-major) ----------
__global__ void pack_wtb_kernel(const float* __restrict__ tsn_w, __bf16* __restrict__ Wtb) {
  int i = blockIdx.x * 256 + threadIdx.x;
  if (i >= NL * NK3) return;
  int l = i / NK3, k = i - l * NK3;
  int s = k >> 12, c = k & (NF - 1);
  Wtb[i] = tobf(tsn_w[(l * NF + c) * 3 + s]);
}

// ---------- pad cs_w0 (256,257) -> bf16 (256,288) ----------
__global__ void pad_csw0_kernel(const float* __restrict__ w, __bf16* __restrict__ wp) {
  int i = blockIdx.x * 256 + threadIdx.x;
  if (i >= 256 * 288) return;
  int n = i / 288, c = i - n * 288;
  wp[i] = tobf((c < 257) ? w[n * 257 + c] : 0.0f);
}

// ---------- Conv1d-as-GEMM via bf16 WMMA; B staged through LDS (async on gfx1250) ----------
__global__ void __launch_bounds__(256) conv_wmma_kernel(
    const __bf16* __restrict__ xbf, const __bf16* __restrict__ Wtb,
    const float* __restrict__ tsn_b, float* __restrict__ feat)
{
  __shared__ __align__(16) __bf16 BS[128 * 32];   // [n_local][k] 32 bf16 per n
  const int tid  = threadIdx.x;
  const int wave = tid >> 5, lane = tid & 31;
  const int l15  = lane & 15, hi = lane >> 4;
  const int m0 = blockIdx.x * 128 + wave * 16;
  const int n0 = blockIdx.y * 128;
  const int row = m0 + l15;
  const int bb = row >> 5, tt = row & 31;

  const __bf16* xrow[3]; bool valid[3];
  #pragma unroll
  for (int s = 0; s < 3; ++s) {
    int st = tt + s - 1;
    valid[s] = (st >= 0) && (st < NT);
    int stc = valid[s] ? st : 0;
    xrow[s] = xbf + (size_t)(bb * NT + stc) * NF + hi * 8;
  }

  const int sn = tid >> 1, sh = tid & 1;              // staging: n = sn, K-half = sh
  const __bf16* wg = Wtb + (size_t)(n0 + sn) * NK3 + sh * 16;
  __bf16* ls = &BS[sn * 32 + sh * 16];

  v8f zero = {};
  v8f acc[8];
  #pragma unroll
  for (int i = 0; i < 8; ++i) acc[i] = zero;

  for (int s = 0; s < 3; ++s) {
    const __bf16* ax = xrow[s];
    const bool av = valid[s];
    for (int kb = 0; kb < NF; kb += 32) {
      __syncthreads();
      const __bf16* g = wg + s * NF + kb;
#if USE_ASYNC_LDS
      __builtin_amdgcn_global_load_async_to_lds_b128(
          (AS1 i4v*)(void*)g, (AS3 i4v*)(void*)ls, 0, 0);
      __builtin_amdgcn_global_load_async_to_lds_b128(
          (AS1 i4v*)(void*)(g + 8), (AS3 i4v*)(void*)(ls + 8), 0, 0);
      __builtin_amdgcn_s_wait_asynccnt(0);
      __syncthreads();
#else
      *(uint4*)ls       = *(const uint4*)g;
      *(uint4*)(ls + 8) = *(const uint4*)(g + 8);
      __syncthreads();
#endif

      union { uint4 q[2]; v16bf v; } ua;
      if (av) {
        if (kb + 1056 < NF) __builtin_prefetch(ax + kb + 1024, 0, 1);
        ua.q[0] = *(const uint4*)(ax + kb);        // K = kb + hi*8 + 0..7
        ua.q[1] = *(const uint4*)(ax + kb + 16);   // K = kb + 16 + hi*8 + 0..7
      } else {
        ua.q[0] = make_uint4(0u, 0u, 0u, 0u);
        ua.q[1] = make_uint4(0u, 0u, 0u, 0u);
      }

      #pragma unroll
      for (int ntl = 0; ntl < 8; ++ntl) {
        const __bf16* lp = &BS[(ntl * 16 + l15) * 32 + hi * 8];
        union { uint4 q[2]; v16bf v; } ub;
        ub.q[0] = *(const uint4*)lp;
        ub.q[1] = *(const uint4*)(lp + 16);
        acc[ntl] = __builtin_amdgcn_wmma_f32_16x16x32_bf16(
            false, ua.v, false, ub.v, (short)0, acc[ntl], false, false);
      }
    }
  }

  const int mb = m0 + hi * 8;
  #pragma unroll
  for (int ntl = 0; ntl < 8; ++ntl) {
    const int n = n0 + ntl * 16 + l15;
    const float bv = tsn_b[n];
    float* cp = feat + (size_t)mb * NL + n;
    #pragma unroll
    for (int r = 0; r < 8; ++r)
      cp[(size_t)r * NL] = acc[ntl][r] + bv;
  }
}

// ---------- generic C = act(A @ W^T + bias), bf16 operands, f32 or bf16 out ----------
// ACT: 0 none, 1 relu, 2 tanh, 3 sigmoid. M,N mult of 16; K mult of 32.
template<int ACT, int OBF>
__global__ void __launch_bounds__(256) gemm_bf16_kernel(
    const __bf16* __restrict__ A, int lda,
    const __bf16* __restrict__ W, int ldw,
    const float* __restrict__ bias,
    void* __restrict__ Cv, int ldc,
    int M, int N, int K)
{
  const int gwave = (blockIdx.x * 256 + threadIdx.x) >> 5;
  const int lane = threadIdx.x & 31;
  const int ntiles = N >> 4;
  const int total = (M >> 4) * ntiles;
  if (gwave >= total) return;
  const int mt = gwave / ntiles, nt = gwave - mt * ntiles;
  const int m0 = mt << 4, n0 = nt << 4;
  const int l15 = lane & 15, hi = lane >> 4;
  const __bf16* ap = A + (size_t)(m0 + l15) * lda + hi * 8;
  const __bf16* wp = W + (size_t)(n0 + l15) * ldw + hi * 8;
  v8f acc = {};
  for (int kb = 0; kb < K; kb += 32) {
    union { uint4 q[2]; v16bf v; } ua, ub;
    ua.q[0] = *(const uint4*)(ap + kb);
    ua.q[1] = *(const uint4*)(ap + kb + 16);
    ub.q[0] = *(const uint4*)(wp + kb);
    ub.q[1] = *(const uint4*)(wp + kb + 16);
    acc = __builtin_amdgcn_wmma_f32_16x16x32_bf16(
        false, ua.v, false, ub.v, (short)0, acc, false, false);
  }
  const float bv = bias[n0 + l15];
  #pragma unroll
  for (int r = 0; r < 8; ++r) {
    float v = acc[r] + bv;
    if (ACT == 1) v = fmaxf(v, 0.0f);
    if (ACT == 2) v = tanhf(v);
    if (ACT == 3) v = sigm(v);
    size_t idx = (size_t)(m0 + hi * 8 + r) * ldc + n0 + l15;
    if (OBF) ((__bf16*)Cv)[idx] = tobf(v);
    else     ((float*)Cv)[idx] = v;
  }
}

// ---------- gated attention logit: A[bt] = sum_j tanhV*sigU*gw[j] + gb ----------
__global__ void gate_kernel(const float* __restrict__ AV, const float* __restrict__ AU,
                            const float* __restrict__ gw, const float* __restrict__ gb,
                            float* __restrict__ Avec, float* __restrict__ At) {
  int i = blockIdx.x * 256 + threadIdx.x;
  if (i >= NBT) return;
  float s = gb[0];
  const float* pv = AV + (size_t)i * NL;
  const float* pu = AU + (size_t)i * NL;
  for (int j = 0; j < NL; ++j) s += pv[j] * pu[j] * gw[j];
  Avec[i] = s; At[i] = s;
}

// ---------- softmax over T (one wave32 per b, lane == t) ----------
__global__ void softmax_kernel(const float* __restrict__ Avec, float* __restrict__ attw) {
  int b = blockIdx.x, t = threadIdx.x;
  float v = Avec[b * NT + t];
  float m = v;
  #pragma unroll
  for (int o = 16; o; o >>= 1) m = fmaxf(m, __shfl_xor(m, o, 32));
  float e = __expf(v - m);
  float s = e;
  #pragma unroll
  for (int o = 16; o; o >>= 1) s += __shfl_xor(s, o, 32);
  attw[b * NT + t] = e / s;
}

// ---------- bag[b,l] = sum_t attw[b,t]*feat[b,t,l] ----------
__global__ void bag_kernel(const float* __restrict__ attw, const float* __restrict__ feat,
                           float* __restrict__ bag) {
  int i = blockIdx.x * 256 + threadIdx.x;
  if (i >= NB * NL) return;
  int b = i >> 8, l = i & 255;
  float s = 0.0f;
  for (int t = 0; t < NT; ++t) s += attw[b * NT + t] * feat[(size_t)(b * NT + t) * NL + l];
  bag[i] = s;
}

// ---------- small final layers ----------
__global__ void bag_head_kernel(const float* __restrict__ h2, const float* __restrict__ w,
                                const float* __restrict__ bb, float* __restrict__ outpair) {
  int i = blockIdx.x * 256 + threadIdx.x;
  if (i >= NB) return;
  float s = bb[0];
  for (int j = 0; j < 32; ++j) s += h2[i * 32 + j] * w[j];
  outpair[i * 2 + 0] = 0.5f * sigm(s);
}
__global__ void seg_head_kernel(const float* __restrict__ s2, const float* __restrict__ w,
                                const float* __restrict__ bb, float* __restrict__ outseg) {
  int i = blockIdx.x * 256 + threadIdx.x;
  if (i >= NBT) return;
  float s = bb[0];
  for (int j = 0; j < 32; ++j) s += s2[i * 32 + j] * w[j];
  outseg[i] = sigm(s);
}

// ---------- seg input build (bf16): [feature | A | pad..288] ----------
__global__ void segin_kernel(const __bf16* __restrict__ featbf, const float* __restrict__ Avec,
                             __bf16* __restrict__ segin) {
  int i = blockIdx.x * 256 + threadIdx.x;
  if (i >= NBT * 288) return;
  int bt = i / 288, c = i - bt * 288;
  segin[i] = (c < NL) ? featbf[(size_t)bt * NL + c]
                      : ((c == NL) ? tobf(Avec[bt]) : tobf(0.0f));
}

// ---------- per-sample 2-means cosine clustering (one block per b, all in LDS) ----------
__global__ void __launch_bounds__(256) kmeans_kernel(
    const float* __restrict__ feat, const float* __restrict__ outseg,
    float* __restrict__ assigns_out, float* __restrict__ outpair)
{
  __shared__ float F[NT * NL];
  __shared__ float FN[NT * NL];
  __shared__ float C0[2 * NL];
  __shared__ float CN[2 * NL];
  __shared__ float NRM[NT];
  __shared__ float SIM[64];
  __shared__ int   ASG[NT];
  __shared__ float CNT[2];
  __shared__ float CNRM[2];
  const int b = blockIdx.x, tid = threadIdx.x;
  const float* fb = feat + (size_t)b * NT * NL;
  for (int i = tid; i < NT * NL; i += 256) F[i] = fb[i];
  __syncthreads();
  if (tid < NT) {
    float s = 0.0f;
    for (int j = 0; j < NL; ++j) { float v = F[tid * NL + j]; s += v * v; }
    NRM[tid] = sqrtf(s) + 1e-8f;
  }
  __syncthreads();
  for (int i = tid; i < NT * NL; i += 256) FN[i] = F[i] / NRM[i >> 8];
  C0[tid]      = F[0 * NL + tid];       // center0 = f[0]
  C0[NL + tid] = F[16 * NL + tid];      // center1 = f[T/2]
  __syncthreads();

  for (int it = 0; it < 100; ++it) {
    if (tid < 2) {
      float s = 0.0f;
      for (int j = 0; j < NL; ++j) { float v = C0[tid * NL + j]; s += v * v; }
      CNRM[tid] = sqrtf(s) + 1e-8f;
    }
    __syncthreads();
    CN[tid]      = C0[tid] / CNRM[0];
    CN[NL + tid] = C0[NL + tid] / CNRM[1];
    __syncthreads();
    if (tid < 64) {
      int t = tid & 31, c = tid >> 5;
      float s = 0.0f;
      for (int j = 0; j < NL; ++j) s += FN[t * NL + j] * CN[c * NL + j];
      SIM[c * 32 + t] = s;
    }
    __syncthreads();
    if (tid < NT) ASG[tid] = (SIM[32 + tid] > SIM[tid]) ? 1 : 0;
    __syncthreads();
    if (tid == 0) {
      float c0 = 0.0f;
      for (int t = 0; t < NT; ++t) c0 += (ASG[t] == 0) ? 1.0f : 0.0f;
      CNT[0] = c0; CNT[1] = (float)NT - c0;
    }
    __syncthreads();
    {
      int j = tid;
      float s0 = 0.0f, s1 = 0.0f;
      for (int t = 0; t < NT; ++t) {
        float v = F[t * NL + j];
        if (ASG[t] == 0) s0 += v; else s1 += v;
      }
      float n0 = CNT[0], n1 = CNT[1];
      float c0n = (n0 > 0.0f) ? s0 / fmaxf(n0, 1.0f) : C0[j];
      float c1n = (n1 > 0.0f) ? s1 / fmaxf(n1, 1.0f) : C0[NL + j];
      C0[j] = c0n; C0[NL + j] = c1n;
    }
    __syncthreads();
  }

  if (tid < NT) assigns_out[b * NT + tid] = (float)ASG[tid];
  if (tid == 0) {
    float s0 = 0.0f, s1 = 0.0f; int n0 = 0;
    for (int t = 0; t < NT; ++t) {
      float v = outseg[b * NT + t];
      if (ASG[t] == 0) { s0 += v; ++n0; } else s1 += v;
    }
    int n1 = NT - n0;
    float m0 = (n0 > 0) ? s0 / (float)n0 : -__builtin_inff();
    float m1 = (n1 > 0) ? s1 / (float)n1 : -__builtin_inff();
    outpair[b * 2 + 1] = 0.5f * fmaxf(m0, m1);
  }
}

// ---------- host side ----------
template<int ACT, int OBF>
static void launch_gemm(const __bf16* A, int lda, const __bf16* W, int ldw, const float* bias,
                        void* C, int ldc, int M, int N, int K, hipStream_t s) {
  int tiles = (M / 16) * (N / 16);
  int blocks = (tiles + 7) / 8;
  gemm_bf16_kernel<ACT, OBF><<<blocks, 256, 0, s>>>(A, lda, W, ldw, bias, C, ldc, M, N, K);
}
static void cast_buf(const float* src, __bf16* dst, int n, hipStream_t s) {
  cast_f2b_kernel<0><<<(n / 4 + 255) / 256, 256, 0, s>>>(src, dst, n);
}

extern "C" void kernel_launch(void* const* d_in, const int* in_sizes, int n_in,
                              void* d_out, int out_size, void* d_ws, size_t ws_size,
                              hipStream_t stream) {
  (void)in_sizes; (void)n_in; (void)out_size; (void)ws_size;
  const float* x      = (const float*)d_in[0];
  const float* tsn_w  = (const float*)d_in[1];
  const float* tsn_b  = (const float*)d_in[2];
  const float* aV_w   = (const float*)d_in[3];
  const float* aV_b   = (const float*)d_in[4];
  const float* aU_w   = (const float*)d_in[5];
  const float* aU_b   = (const float*)d_in[6];
  const float* gate_w = (const float*)d_in[7];
  const float* gate_b = (const float*)d_in[8];
  const float* cb_w0  = (const float*)d_in[9];
  const float* cb_b0  = (const float*)d_in[10];
  const float* cb_w1  = (const float*)d_in[11];
  const float* cb_b1  = (const float*)d_in[12];
  const float* cb_w2  = (const float*)d_in[13];
  const float* cb_b2  = (const float*)d_in[14];
  const float* cb_w3  = (const float*)d_in[15];
  const float* cb_b3  = (const float*)d_in[16];
  const float* cs_w0  = (const float*)d_in[17];
  const float* cs_b0  = (const float*)d_in[18];
  const float* cs_w1  = (const float*)d_in[19];
  const float* cs_b1  = (const float*)d_in[20];
  const float* cs_w2  = (const float*)d_in[21];
  const float* cs_b2  = (const float*)d_in[22];
  const float* cs_w3  = (const float*)d_in[23];
  const float* cs_b3  = (const float*)d_in[24];

  float* out     = (float*)d_out;
  float* feat    = out;                          // (B,T,L)
  float* assigns = feat + (size_t)NBT * NL;      // (B,T)
  float* outseg  = assigns + NBT;                // (B,T)
  float* outpair = outseg + NBT;                 // (B,2)
  float* At      = outpair + 2 * NB;             // (B,1,T)

  char* wsb = (char*)d_ws;
  size_t off = 0;
  auto take = [&](size_t bytes) -> void* {
    void* p = wsb + off;
    off = (off + bytes + 255) & ~(size_t)255;
    return p;
  };
  __bf16* xbf    = (__bf16*)take((size_t)NBT * NF * 2);
  __bf16* Wtb    = (__bf16*)take((size_t)NL * NK3 * 2);
  __bf16* featbf = (__bf16*)take((size_t)NBT * NL * 2);
  __bf16* aVwb   = (__bf16*)take((size_t)NL * NL * 2);
  __bf16* aUwb   = (__bf16*)take((size_t)NL * NL * 2);
  __bf16* cbw0b  = (__bf16*)take((size_t)256 * 256 * 2);
  __bf16* cbw1b  = (__bf16*)take((size_t)128 * 256 * 2);
  __bf16* cbw2b  = (__bf16*)take((size_t)32 * 128 * 2);
  __bf16* csw0pb = (__bf16*)take((size_t)256 * 288 * 2);
  __bf16* csw1b  = (__bf16*)take((size_t)128 * 256 * 2);
  __bf16* csw2b  = (__bf16*)take((size_t)32 * 128 * 2);
  float*  AV     = (float*)take((size_t)NBT * NL * 4);
  float*  AU     = (float*)take((size_t)NBT * NL * 4);
  float*  Avec   = (float*)take((size_t)NBT * 4);
  float*  attw   = (float*)take((size_t)NBT * 4);
  float*  bag    = (float*)take((size_t)NB * NL * 4);
  __bf16* bagbf  = (__bf16*)take((size_t)NB * NL * 2);
  __bf16* h0b    = (__bf16*)take((size_t)NB * 256 * 2);
  __bf16* h1b    = (__bf16*)take((size_t)NB * 128 * 2);
  float*  h2     = (float*)take((size_t)NB * 32 * 4);
  __bf16* seginb = (__bf16*)take((size_t)NBT * 288 * 2);
  __bf16* s0b    = (__bf16*)take((size_t)NBT * 256 * 2);
  __bf16* s1b    = (__bf16*)take((size_t)NBT * 128 * 2);
  float*  s2     = (float*)take((size_t)NBT * 32 * 4);

  // one-pass conversions / packing
  cast_f2b_kernel<1><<<(NBT * NF / 4 + 255) / 256, 256, 0, stream>>>(x, xbf, NBT * NF);
  pack_wtb_kernel<<<(NL * NK3 + 255) / 256, 256, 0, stream>>>(tsn_w, Wtb);
  pad_csw0_kernel<<<(256 * 288 + 255) / 256, 256, 0, stream>>>(cs_w0, csw0pb);
  cast_buf(aV_w, aVwb, NL * NL, stream);
  cast_buf(aU_w, aUwb, NL * NL, stream);
  cast_buf(cb_w0, cbw0b, 256 * 256, stream);
  cast_buf(cb_w1, cbw1b, 128 * 256, stream);
  cast_buf(cb_w2, cbw2b, 32 * 128, stream);
  cast_buf(cs_w1, csw1b, 128 * 256, stream);
  cast_buf(cs_w2, csw2b, 32 * 128, stream);

  // conv (dominant GEMM)
  conv_wmma_kernel<<<dim3(NBT / 128, 2), 256, 0, stream>>>(xbf, Wtb, tsn_b, feat);
  cast_buf(feat, featbf, NBT * NL, stream);

  // gated attention
  launch_gemm<2, 0>(featbf, NL, aVwb, NL, aV_b, AV, NL, NBT, NL, NL, stream);   // tanh
  launch_gemm<3, 0>(featbf, NL, aUwb, NL, aU_b, AU, NL, NBT, NL, NL, stream);   // sigmoid
  gate_kernel<<<(NBT + 255) / 256, 256, 0, stream>>>(AV, AU, gate_w, gate_b, Avec, At);
  softmax_kernel<<<NB, 32, 0, stream>>>(Avec, attw);
  bag_kernel<<<(NB * NL + 255) / 256, 256, 0, stream>>>(attw, feat, bag);
  cast_buf(bag, bagbf, NB * NL, stream);

  // bag classifier MLP (bf16 chain)
  launch_gemm<1, 1>(bagbf, 256, cbw0b, 256, cb_b0, h0b, 256, NB, 256, 256, stream);
  launch_gemm<1, 1>(h0b, 256, cbw1b, 256, cb_b1, h1b, 128, NB, 128, 256, stream);
  launch_gemm<1, 0>(h1b, 128, cbw2b, 128, cb_b2, h2, 32, NB, 32, 128, stream);
  bag_head_kernel<<<(NB + 255) / 256, 256, 0, stream>>>(h2, cb_w3, cb_b3, outpair);

  // seg classifier MLP (K padded 257 -> 288, bf16 chain)
  segin_kernel<<<(NBT * 288 + 255) / 256, 256, 0, stream>>>(featbf, Avec, seginb);
  launch_gemm<1, 1>(seginb, 288, csw0pb, 288, cs_b0, s0b, 256, NBT, 256, 288, stream);
  launch_gemm<1, 1>(s0b, 256, csw1b, 256, cs_b1, s1b, 128, NBT, 128, 256, stream);
  launch_gemm<1, 0>(s1b, 128, csw2b, 128, cs_b2, s2, 32, NBT, 32, 128, stream);
  seg_head_kernel<<<(NBT + 255) / 256, 256, 0, stream>>>(s2, cs_w3, cs_b3, outseg);

  // per-sample kmeans + output2
  kmeans_kernel<<<NB, 256, 0, stream>>>(feat, outseg, assigns, outpair);
}
